// basic_block_90675349553370
// MI455X (gfx1250) — matrix operand order repaired and checked by reading, not compiled, not run on hardware
//
#include <hip/hip_runtime.h>
#include <hip/hip_bf16.h>
#include <math.h>

typedef __attribute__((ext_vector_type(16))) _Float16 v16h;
typedef __attribute__((ext_vector_type(8)))  _Float16 v8h;
typedef __attribute__((ext_vector_type(8)))  float    v8f;

// ---- problem constants (from reference) ----
constexpr int Bc = 4, Hc = 128, Wc = 128, Lc = Hc * Wc;   // pixel tokens
constexpr int Ec = 288, NHh = 6, HDc = 48;                // embed, heads, head dim
constexpr int H2c = 64, W2c = 64, L2c = H2c * W2c;        // region tokens
constexpr int KPAD = 64;                                  // padded K for 48-dim GEMMs

// =====================================================================
// f32 -> f16 convert (plain)
// =====================================================================
__global__ void k_f2h(const float* __restrict__ s, _Float16* __restrict__ d, size_t n) {
  size_t i = blockIdx.x * (size_t)blockDim.x + threadIdx.x;
  if (i < n) d[i] = (_Float16)s[i];
}

// f32 (rows x 48) -> f16 (rows x 64), zero padded
__global__ void k_f2h_pad(const float* __restrict__ s, _Float16* __restrict__ d, int rows) {
  int i = blockIdx.x * blockDim.x + threadIdx.x;
  int total = rows * KPAD;
  if (i >= total) return;
  int c = i % KPAD, r = i / KPAD;
  d[i] = (c < 48) ? (_Float16)s[r * 48 + c] : (_Float16)0.f;
}

// =====================================================================
// (optional LayerNorm) + mamba in_proj; writes xs/z channel-major (B,48,L)
// block = 96 threads, grid = B*L
// =====================================================================
__global__ void k_ln_inproj(const float* __restrict__ x, const float* __restrict__ ln_w,
                            const float* __restrict__ ln_b, int useLN,
                            const float* __restrict__ w /*96x48*/,
                            float* __restrict__ xs, float* __restrict__ z, int L) {
  int tok = blockIdx.x;
  int b = tok / L, l = tok - b * L;
  int t = threadIdx.x;  // 0..95
  __shared__ float xn[48];
  __shared__ float red[2];
  if (t < 48) xn[t] = x[(size_t)tok * 48 + t];
  __syncthreads();
  if (useLN) {
    if (t == 0) {
      float s = 0.f;
      for (int c = 0; c < 48; c++) s += xn[c];
      float mu = s / 48.f, s2 = 0.f;
      for (int c = 0; c < 48; c++) { float d0 = xn[c] - mu; s2 += d0 * d0; }
      red[0] = mu; red[1] = rsqrtf(s2 / 48.f + 1e-5f);
    }
    __syncthreads();
    if (t < 48) xn[t] = (xn[t] - red[0]) * red[1] * ln_w[t] + ln_b[t];
    __syncthreads();
  }
  float acc = 0.f;
  const float* wr = w + t * 48;
  #pragma unroll
  for (int k = 0; k < 48; k++) acc += wr[k] * xn[k];
  if (t < 48) xs[((size_t)b * 48 + t) * L + l] = acc;
  else        z [((size_t)b * 48 + (t - 48)) * L + l] = acc;
}

// =====================================================================
// depthwise conv1d (k=3, same pad) + SiLU, channel-major
// =====================================================================
__global__ void k_dwconv_silu(const float* __restrict__ in, const float* __restrict__ w,
                              float* __restrict__ out, int L) {
  size_t i = blockIdx.x * (size_t)blockDim.x + threadIdx.x;
  size_t total = (size_t)Bc * 48 * L;
  if (i >= total) return;
  int l = (int)(i % L);
  size_t bc = i / L;
  int c = (int)(bc % 48);
  const float* row = in + bc * L;
  float acc = 0.f;
  #pragma unroll
  for (int t = 0; t < 3; t++) {
    int ll = l + t - 1;
    if (ll >= 0 && ll < L) acc += w[c * 3 + t] * row[ll];
  }
  out[i] = acc / (1.f + expf(-acc));  // silu
}

// =====================================================================
// x_proj (35x48) + dt_proj (48x3); block = 48 threads, grid = B*L
// =====================================================================
__global__ void k_xproj(const float* __restrict__ xs, const float* __restrict__ xpw,
                        const float* __restrict__ dtw,
                        float* __restrict__ dt, float* __restrict__ Bm,
                        float* __restrict__ Cm, int L) {
  int tok = blockIdx.x;
  int b = tok / L, l = tok - b * L;
  int t = threadIdx.x;  // 0..47
  __shared__ float xl[48];
  __shared__ float xd[35];
  xl[t] = xs[((size_t)b * 48 + t) * L + l];
  __syncthreads();
  if (t < 35) {
    float a = 0.f;
    const float* wr = xpw + t * 48;
    #pragma unroll
    for (int k = 0; k < 48; k++) a += wr[k] * xl[k];
    xd[t] = a;
  }
  __syncthreads();
  {
    float a = 0.f;
    #pragma unroll
    for (int r = 0; r < 3; r++) a += dtw[t * 3 + r] * xd[r];
    dt[((size_t)b * 48 + t) * L + l] = a;
  }
  if (t < 16) {
    Bm[((size_t)b * 16 + t) * L + l] = xd[3 + t];
    Cm[((size_t)b * 16 + t) * L + l] = xd[19 + t];
  }
}

// =====================================================================
// selective scan: one wave per (b,d); 16 state lanes; shfl_xor reduce
// =====================================================================
__global__ void k_scan(const float* __restrict__ u, const float* __restrict__ dt,
                       const float* __restrict__ A_log, const float* __restrict__ Dp,
                       const float* __restrict__ dtb,
                       const float* __restrict__ Bm, const float* __restrict__ Cm,
                       float* __restrict__ y, int L) {
  int bd = blockIdx.x;
  int b = bd / 48, d = bd - b * 48;
  int n = threadIdx.x;  // wave32; lanes 0..15 carry state
  float A = (n < 16) ? -expf(A_log[d * 16 + n]) : 0.f;
  float bias = dtb[d], Dd = Dp[d];
  const float* urow = u + ((size_t)b * 48 + d) * L;
  const float* drow = dt + ((size_t)b * 48 + d) * L;
  const float* Brow = Bm + ((size_t)b * 16 + (n < 16 ? n : 0)) * L;
  const float* Crow = Cm + ((size_t)b * 16 + (n < 16 ? n : 0)) * L;
  float h = 0.f;
  for (int l = 0; l < L; l++) {
    float delta = drow[l] + bias;
    delta = (delta > 20.f) ? delta : log1pf(expf(delta));
    float uu = urow[l];
    float p = 0.f;
    if (n < 16) {
      h = expf(delta * A) * h + delta * uu * Brow[l];
      p = h * Crow[l];
    }
    p += __shfl_xor(p, 1);
    p += __shfl_xor(p, 2);
    p += __shfl_xor(p, 4);
    p += __shfl_xor(p, 8);
    if (n == 0) y[((size_t)b * 48 + d) * L + l] = p + uu * Dd;
  }
}

// =====================================================================
// out_proj: concat(y,z) @ W^T (48x96); token-major f16 out, stride KPAD
// with explicit zero pad so the following GEMM needs no K guards.
// =====================================================================
__global__ void k_outproj(const float* __restrict__ y, const float* __restrict__ z,
                          const float* __restrict__ w, _Float16* __restrict__ outh, int L) {
  int tok = blockIdx.x;
  int b = tok / L, l = tok - b * L;
  int t = threadIdx.x;  // 0..47
  __shared__ float cat[96];
  cat[t]      = y[((size_t)b * 48 + t) * L + l];
  cat[48 + t] = z[((size_t)b * 48 + t) * L + l];
  __syncthreads();
  float a = 0.f;
  const float* wr = w + t * 96;
  #pragma unroll
  for (int k = 0; k < 96; k++) a += wr[k] * cat[k];
  outh[(size_t)tok * KPAD + t] = (_Float16)a;
  if (t < 16) outh[(size_t)tok * KPAD + 48 + t] = (_Float16)0.f;  // K pad
}

// =====================================================================
// WMMA GEMM: C[M,N] = A[M,K] * W[N,K]^T + bias.  f16 in, f32 acc.
// K, output mode are template parameters -> no guards/branches anywhere.
// Wave tile: 16(M) x 48(N) (3 fragments); 4 waves/block on separate M tiles.
// Grid: (M/64, N/48).  N must be a multiple of 48 (288, 864, 48 all are).
// Fragment layouts per CDNA5 ISA 7.12.2 (wave32):
//   A 16x32 f16 : lane half h, row mr; two contiguous 16B runs at k0+16g+8h
//   B 32x16 f16 : col mr; two contiguous 16B runs at k0+16g+8h
//   C/D f32     : m = 8h + r, n = mr
// =====================================================================
template <int K, bool F32OUT, bool F16OUT>
__global__ void k_gemm_h(const _Float16* __restrict__ A, const _Float16* __restrict__ Wt,
                         const float* __restrict__ bias,
                         float* __restrict__ Cf, _Float16* __restrict__ Ch,
                         int M, int N) {
  int lane = threadIdx.x & 31;
  int wave = threadIdx.x >> 5;
  int m0 = (blockIdx.x * 4 + wave) * 16;
  int nbase = blockIdx.y * 48;
  int h = lane >> 4;    // lane half
  int mr = lane & 15;   // A row / B-C column within tile
  v8f z8 = {0.f, 0.f, 0.f, 0.f, 0.f, 0.f, 0.f, 0.f};
  v8f acc0 = z8, acc1 = z8, acc2 = z8;
  const _Float16* Arow = A + (size_t)(m0 + mr) * K;
  const _Float16* W0 = Wt + (size_t)(nbase + mr) * K;
  const _Float16* W1 = W0 + (size_t)16 * K;
  const _Float16* W2 = W0 + (size_t)32 * K;
  #pragma unroll
  for (int k0 = 0; k0 < K; k0 += 32) {
    __builtin_prefetch(Arow + k0 + 256, 0, 1);  // global_prefetch_b8
    const v8h a0 = *(const v8h*)(Arow + k0 + h * 8);
    const v8h a1 = *(const v8h*)(Arow + k0 + 16 + h * 8);
    v16h a = __builtin_shufflevector(a0, a1, 0, 1, 2, 3, 4, 5, 6, 7,
                                     8, 9, 10, 11, 12, 13, 14, 15);
    {
      const v8h b0 = *(const v8h*)(W0 + k0 + h * 8);
      const v8h b1 = *(const v8h*)(W0 + k0 + 16 + h * 8);
      v16h bf = __builtin_shufflevector(b0, b1, 0, 1, 2, 3, 4, 5, 6, 7,
                                        8, 9, 10, 11, 12, 13, 14, 15);
      acc0 = __builtin_amdgcn_wmma_f32_16x16x32_f16(false, a, false, bf, (short)0,
                                                    acc0, false, false);
    }
    {
      const v8h b0 = *(const v8h*)(W1 + k0 + h * 8);
      const v8h b1 = *(const v8h*)(W1 + k0 + 16 + h * 8);
      v16h bf = __builtin_shufflevector(b0, b1, 0, 1, 2, 3, 4, 5, 6, 7,
                                        8, 9, 10, 11, 12, 13, 14, 15);
      acc1 = __builtin_amdgcn_wmma_f32_16x16x32_f16(false, a, false, bf, (short)0,
                                                    acc1, false, false);
    }
    {
      const v8h b0 = *(const v8h*)(W2 + k0 + h * 8);
      const v8h b1 = *(const v8h*)(W2 + k0 + 16 + h * 8);
      v16h bf = __builtin_shufflevector(b0, b1, 0, 1, 2, 3, 4, 5, 6, 7,
                                        8, 9, 10, 11, 12, 13, 14, 15);
      acc2 = __builtin_amdgcn_wmma_f32_16x16x32_f16(false, a, false, bf, (short)0,
                                                    acc2, false, false);
    }
  }
  // branch-free epilogue: per-lane base for row m0+8h, column nbase+mr
  v8f accs[3] = {acc0, acc1, acc2};
  size_t rowbase = (size_t)(m0 + h * 8) * N + nbase + mr;
  #pragma unroll
  for (int nf = 0; nf < 3; nf++) {
    float bsv = bias[nbase + nf * 16 + mr];
    #pragma unroll
    for (int r = 0; r < 8; r++) {
      size_t idx = rowbase + (size_t)r * N + nf * 16;
      float val = accs[nf][r] + bsv;
      if (F32OUT) Cf[idx] = val;
      if (F16OUT) Ch[idx] = (_Float16)val;
    }
  }
}

// =====================================================================
// Neighborhood attention, 3x3 window, thread per (b,i,j,head)
// q/k/v rows are 16B-aligned f16 -> v8h (b128) loads
// =====================================================================
__global__ void k_nat(const _Float16* __restrict__ qkv, _Float16* __restrict__ outh,
                      int Hh, int Ww) {
  int idx = blockIdx.x * blockDim.x + threadIdx.x;
  int total = Bc * Hh * Ww * NHh;
  if (idx >= total) return;
  int head = idx % NHh; int r = idx / NHh;
  int j = r % Ww; r /= Ww;
  int i = r % Hh; int b = r / Hh;
  size_t base = ((size_t)(b * Hh + i) * Ww + j) * (3 * Ec) + head * 48;
  const float scale = rsqrtf((float)HDc);
  float q[48];
  #pragma unroll
  for (int g = 0; g < 6; g++) {
    v8h qv = *(const v8h*)(qkv + base + g * 8);
    #pragma unroll
    for (int e = 0; e < 8; e++) q[g * 8 + e] = (float)qv[e] * scale;
  }
  float logit[9];
  float mx = -1e30f;
  for (int di = 0; di < 3; di++)
    for (int dj = 0; dj < 3; dj++) {
      int ii = i + di - 1, jj = j + dj - 1;
      float lg;
      if (ii >= 0 && ii < Hh && jj >= 0 && jj < Ww) {
        const _Float16* kp =
            qkv + ((size_t)(b * Hh + ii) * Ww + jj) * (3 * Ec) + Ec + head * 48;
        float a = 0.f;
        #pragma unroll
        for (int g = 0; g < 6; g++) {
          v8h kv = *(const v8h*)(kp + g * 8);
          #pragma unroll
          for (int e = 0; e < 8; e++) a += q[g * 8 + e] * (float)kv[e];
        }
        lg = a;
      } else lg = -1e30f;
      logit[di * 3 + dj] = lg;
      mx = fmaxf(mx, lg);
    }
  float ssum = 0.f;
  #pragma unroll
  for (int t = 0; t < 9; t++) { float e = expf(logit[t] - mx); logit[t] = e; ssum += e; }
  float inv = 1.f / ssum;
  float o[48];
  #pragma unroll
  for (int c = 0; c < 48; c++) o[c] = 0.f;
  for (int di = 0; di < 3; di++)
    for (int dj = 0; dj < 3; dj++) {
      int ii = i + di - 1, jj = j + dj - 1;
      if (ii < 0 || ii >= Hh || jj < 0 || jj >= Ww) continue;
      float wv = logit[di * 3 + dj] * inv;
      const _Float16* vp =
          qkv + ((size_t)(b * Hh + ii) * Ww + jj) * (3 * Ec) + 2 * Ec + head * 48;
      #pragma unroll
      for (int g = 0; g < 6; g++) {
        v8h vv = *(const v8h*)(vp + g * 8);
        #pragma unroll
        for (int e = 0; e < 8; e++) o[g * 8 + e] += wv * (float)vv[e];
      }
    }
  size_t ob = ((size_t)(b * Hh + i) * Ww + j) * Ec + head * 48;
  #pragma unroll
  for (int c = 0; c < 48; c++) outh[ob + c] = (_Float16)o[c];
}

// =====================================================================
// LN2 + pixel_unshuffle(2) + 1x1 conv (192->48) -> region tokens (B,L2,48)
// =====================================================================
__global__ void k_down(const float* __restrict__ x, const float* __restrict__ ln_w,
                       const float* __restrict__ ln_b,
                       const float* __restrict__ w /*48x192*/, const float* __restrict__ bvec,
                       float* __restrict__ out) {
  int tok = blockIdx.x;
  int b = tok / L2c, l2 = tok - b * L2c;
  int i2 = l2 / W2c, j2 = l2 - i2 * W2c;
  int t = threadIdx.x;  // 0..47
  __shared__ float tl[192];
  __shared__ float red[2];
  #pragma unroll
  for (int s = 0; s < 4; s++) {
    int r1 = s >> 1, r2 = s & 1;
    int l = (2 * i2 + r1) * Wc + (2 * j2 + r2);
    tl[s * 48 + t] = x[((size_t)b * Lc + l) * 48 + t];
  }
  __syncthreads();
  for (int s = 0; s < 4; s++) {
    if (t == 0) {
      float su = 0.f;
      for (int c = 0; c < 48; c++) su += tl[s * 48 + c];
      float mu = su / 48.f, s2 = 0.f;
      for (int c = 0; c < 48; c++) { float d0 = tl[s * 48 + c] - mu; s2 += d0 * d0; }
      red[0] = mu; red[1] = rsqrtf(s2 / 48.f + 1e-5f);
    }
    __syncthreads();
    float nv = (tl[s * 48 + t] - red[0]) * red[1] * ln_w[t] + ln_b[t];
    __syncthreads();
    tl[s * 48 + t] = nv;
    __syncthreads();
  }
  float a = bvec[t];
  const float* wr = w + t * 192;
  #pragma unroll
  for (int cin = 0; cin < 192; cin++) a += wr[cin] * tl[(cin & 3) * 48 + (cin >> 2)];
  out[(size_t)tok * 48 + t] = a;
}

// =====================================================================
// 1x1 conv up (48->192) + pixel_shuffle(2) + 1x1 conv (48->48) -> (B,L,48)
// =====================================================================
__global__ void k_up(const float* __restrict__ xr, const float* __restrict__ bw /*192x48*/,
                     const float* __restrict__ bb, const float* __restrict__ aw /*48x48*/,
                     const float* __restrict__ ab, float* __restrict__ out) {
  int pix = blockIdx.x;
  int b = pix / Lc; int l = pix - b * Lc;
  int i = l / Wc, j = l - i * Wc;
  int t = threadIdx.x;  // 0..47
  __shared__ float src[48];
  __shared__ float mid[48];
  int l2 = (i >> 1) * W2c + (j >> 1);
  src[t] = xr[((size_t)b * L2c + l2) * 48 + t];
  __syncthreads();
  int cc = t * 4 + (i & 1) * 2 + (j & 1);
  float a = bb[cc];
  const float* wr = bw + (size_t)cc * 48;
  #pragma unroll
  for (int k = 0; k < 48; k++) a += wr[k] * src[k];
  mid[t] = a;
  __syncthreads();
  float o = ab[t];
  const float* ar = aw + t * 48;
  #pragma unroll
  for (int k = 0; k < 48; k++) o += ar[k] * mid[k];
  out[(size_t)pix * 48 + t] = o;
}

// =====================================================================
// fusion conv1 3x3 (96->48) + exact GELU; thread per (b,pos,oc)
// =====================================================================
__global__ void k_conv1(const float* __restrict__ xp, const float* __restrict__ xr,
                        const float* __restrict__ w /*48x96x3x3*/, const float* __restrict__ bvec,
                        float* __restrict__ out) {
  int idx = blockIdx.x * blockDim.x + threadIdx.x;
  int total = Bc * Lc * 48;
  if (idx >= total) return;
  int oc = idx % 48; int p = idx / 48;
  int b = p / Lc; int l = p - b * Lc;
  int i = l / Wc, j = l - i * Wc;
  float a = bvec[oc];
  const float* wb = w + (size_t)oc * 96 * 9;
  for (int di = 0; di < 3; di++) {
    int ii = i + di - 1; if (ii < 0 || ii >= Hc) continue;
    for (int dj = 0; dj < 3; dj++) {
      int jj = j + dj - 1; if (jj < 0 || jj >= Wc) continue;
      size_t pb = ((size_t)b * Lc + ii * Wc + jj) * 48;
      const float* wk = wb + di * 3 + dj;
      #pragma unroll
      for (int ci = 0; ci < 48; ci++) a += wk[ci * 9] * xp[pb + ci];
      #pragma unroll
      for (int ci = 0; ci < 48; ci++) a += wk[(48 + ci) * 9] * xr[pb + ci];
    }
  }
  out[idx] = 0.5f * a * (1.f + erff(a * 0.70710678118654752f));
}

// =====================================================================
// conv2 3x3 (48->2) + softmax + blend + final linear + residual
// =====================================================================
__global__ void k_fuse(const float* __restrict__ g, const float* __restrict__ c2w /*2x48x3x3*/,
                       const float* __restrict__ c2b, const float* __restrict__ xp,
                       const float* __restrict__ xr, const float* __restrict__ lw /*48x48*/,
                       const float* __restrict__ lb, const float* __restrict__ xin,
                       float* __restrict__ out) {
  int pix = blockIdx.x;
  int b = pix / Lc; int l = pix - b * Lc;
  int i = l / Wc, j = l - i * Wc;
  int t = threadIdx.x;  // 0..47
  __shared__ float f[2];
  __shared__ float xm[48];
  if (t < 2) {
    float a = c2b[t];
    const float* wb = c2w + (size_t)t * 48 * 9;
    for (int di = 0; di < 3; di++) {
      int ii = i + di - 1; if (ii < 0 || ii >= Hc) continue;
      for (int dj = 0; dj < 3; dj++) {
        int jj = j + dj - 1; if (jj < 0 || jj >= Wc) continue;
        size_t pb = ((size_t)b * Lc + ii * Wc + jj) * 48;
        const float* wk = wb + di * 3 + dj;
        #pragma unroll
        for (int ci = 0; ci < 48; ci++) a += wk[ci * 9] * g[pb + ci];
      }
    }
    f[t] = a;
  }
  __syncthreads();
  float m = fmaxf(f[0], f[1]);
  float e0 = expf(f[0] - m), e1 = expf(f[1] - m);
  float w0 = e0 / (e0 + e1), w1 = e1 / (e0 + e1);
  size_t pb = (size_t)pix * 48;
  xm[t] = w0 * xp[pb + t] + w1 * xr[pb + t];
  __syncthreads();
  float a = lb[t];
  const float* wr = lw + t * 48;
  #pragma unroll
  for (int k = 0; k < 48; k++) a += wr[k] * xm[k];
  out[pb + t] = a + xin[pb + t];
}

// =====================================================================
// host launcher
// =====================================================================
extern "C" void kernel_launch(void* const* d_in, const int* in_sizes, int n_in,
                              void* d_out, int out_size, void* d_ws, size_t ws_size,
                              hipStream_t stream) {
  (void)in_sizes; (void)n_in; (void)out_size; (void)ws_size;
  const float* x      = (const float*)d_in[0];
  const float* ln1_w  = (const float*)d_in[3];
  const float* ln1_b  = (const float*)d_in[4];
  const float* ln2_w  = (const float*)d_in[5];
  const float* ln2_b  = (const float*)d_in[6];
  // mamba_p (7..15)
  const float* mp_inw  = (const float*)d_in[7];
  const float* mp_cxw  = (const float*)d_in[8];
  const float* mp_czw  = (const float*)d_in[9];
  const float* mp_xpw  = (const float*)d_in[10];
  const float* mp_dtw  = (const float*)d_in[11];
  const float* mp_dtb  = (const float*)d_in[12];
  const float* mp_alog = (const float*)d_in[13];
  const float* mp_D    = (const float*)d_in[14];
  const float* mp_opw  = (const float*)d_in[15];
  // mamba_r (16..24)
  const float* mr_inw  = (const float*)d_in[16];
  const float* mr_cxw  = (const float*)d_in[17];
  const float* mr_czw  = (const float*)d_in[18];
  const float* mr_xpw  = (const float*)d_in[19];
  const float* mr_dtw  = (const float*)d_in[20];
  const float* mr_dtb  = (const float*)d_in[21];
  const float* mr_alog = (const float*)d_in[22];
  const float* mr_D    = (const float*)d_in[23];
  const float* mr_opw  = (const float*)d_in[24];
  // nat1 / nat2
  const float* n1_qkvw = (const float*)d_in[25];
  const float* n1_qkvb = (const float*)d_in[26];
  const float* n1_pw   = (const float*)d_in[27];
  const float* n1_pb   = (const float*)d_in[28];
  const float* n2_qkvw = (const float*)d_in[29];
  const float* n2_qkvb = (const float*)d_in[30];
  const float* n2_pw   = (const float*)d_in[31];
  const float* n2_pb   = (const float*)d_in[32];
  const float* lb1_w = (const float*)d_in[33];
  const float* lb1_b = (const float*)d_in[34];
  const float* lb2_w = (const float*)d_in[35];
  const float* lb2_b = (const float*)d_in[36];
  const float* la1_w = (const float*)d_in[37];
  const float* la1_b = (const float*)d_in[38];
  const float* la2_w = (const float*)d_in[39];
  const float* la2_b = (const float*)d_in[40];
  const float* cdw = (const float*)d_in[41];
  const float* cdb = (const float*)d_in[42];
  const float* cbw = (const float*)d_in[43];
  const float* cbb = (const float*)d_in[44];
  const float* caw = (const float*)d_in[45];
  const float* cab = (const float*)d_in[46];
  const float* c1w = (const float*)d_in[47];
  const float* c1b = (const float*)d_in[48];
  const float* c2w = (const float*)d_in[49];
  const float* c2b = (const float*)d_in[50];
  const float* lw  = (const float*)d_in[51];
  const float* lbv = (const float*)d_in[52];
  float* out = (float*)d_out;

  // ---- workspace bump allocator ----
  char* base = (char*)d_ws;
  size_t off = 0;
  auto alloc = [&](size_t bytes) -> void* {
    off = (off + 255) & ~(size_t)255;
    void* p = base + off;
    off += bytes;
    return p;
  };
  const size_t MP = (size_t)Bc * Lc;   // 65536 pixel tokens
  const size_t MR = (size_t)Bc * L2c;  // 16384 region tokens

  float* xs_raw = (float*)alloc(48 * MP * 4);
  float* z_raw  = (float*)alloc(48 * MP * 4);
  float* xs_c   = (float*)alloc(48 * MP * 4);
  float* z_c    = (float*)alloc(48 * MP * 4);
  float* dtbuf  = (float*)alloc(48 * MP * 4);
  float* Bm     = (float*)alloc(16 * MP * 4);
  float* Cm     = (float*)alloc(16 * MP * 4);
  float* ybuf   = (float*)alloc(48 * MP * 4);
  _Float16* tok_h  = (_Float16*)alloc((size_t)KPAD * MP * 2);     // K-padded
  _Float16* e_h    = (_Float16*)alloc((size_t)Ec * MP * 2);
  _Float16* qkv_h  = (_Float16*)alloc((size_t)3 * Ec * MP * 2);
  _Float16* attn_h = (_Float16*)alloc((size_t)Ec * MP * 2);
  _Float16* proj_h = (_Float16*)alloc((size_t)Ec * MP * 2);
  float* xp_fin = (float*)alloc(48 * MP * 4);
  float* xr0    = (float*)alloc(48 * MR * 4);
  float* xr_tok = (float*)alloc(48 * MR * 4);
  float* xr_img = (float*)alloc(48 * MP * 4);
  float* gbuf   = (float*)alloc(48 * MP * 4);
  _Float16* w_lb1  = (_Float16*)alloc((size_t)Ec * KPAD * 2);     // K-padded
  _Float16* w_lb2  = (_Float16*)alloc((size_t)Ec * KPAD * 2);     // K-padded
  _Float16* w_la1  = (_Float16*)alloc((size_t)48 * Ec * 2);
  _Float16* w_la2  = (_Float16*)alloc((size_t)48 * Ec * 2);
  _Float16* w_qkv1 = (_Float16*)alloc((size_t)3 * Ec * Ec * 2);
  _Float16* w_qkv2 = (_Float16*)alloc((size_t)3 * Ec * Ec * 2);
  _Float16* w_pj1  = (_Float16*)alloc((size_t)Ec * Ec * 2);
  _Float16* w_pj2  = (_Float16*)alloc((size_t)Ec * Ec * 2);

  auto cvt = [&](const float* s, _Float16* d, size_t n) {
    k_f2h<<<dim3((unsigned)((n + 255) / 256)), 256, 0, stream>>>(s, d, n);
  };

  // ---- weight conversions (f16 operands for WMMA) ----
  k_f2h_pad<<<(Ec * KPAD + 255) / 256, 256, 0, stream>>>(lb1_w, w_lb1, Ec);
  k_f2h_pad<<<(Ec * KPAD + 255) / 256, 256, 0, stream>>>(lb2_w, w_lb2, Ec);
  cvt(la1_w, w_la1, (size_t)48 * Ec);
  cvt(la2_w, w_la2, (size_t)48 * Ec);
  cvt(n1_qkvw, w_qkv1, (size_t)3 * Ec * Ec);
  cvt(n2_qkvw, w_qkv2, (size_t)3 * Ec * Ec);
  cvt(n1_pw, w_pj1, (size_t)Ec * Ec);
  cvt(n2_pw, w_pj2, (size_t)Ec * Ec);

  // ================= pixel branch =================
  k_ln_inproj<<<(unsigned)MP, 96, 0, stream>>>(x, ln1_w, ln1_b, 1, mp_inw, xs_raw, z_raw, Lc);
  {
    size_t tot = (size_t)Bc * 48 * Lc;
    unsigned g = (unsigned)((tot + 255) / 256);
    k_dwconv_silu<<<g, 256, 0, stream>>>(xs_raw, mp_cxw, xs_c, Lc);
    k_dwconv_silu<<<g, 256, 0, stream>>>(z_raw, mp_czw, z_c, Lc);
  }
  k_xproj<<<(unsigned)MP, 48, 0, stream>>>(xs_c, mp_xpw, mp_dtw, dtbuf, Bm, Cm, Lc);
  k_scan<<<Bc * 48, 32, 0, stream>>>(xs_c, dtbuf, mp_alog, mp_D, mp_dtb, Bm, Cm, ybuf, Lc);
  k_outproj<<<(unsigned)MP, 48, 0, stream>>>(ybuf, z_c, mp_opw, tok_h, Lc);
  // lb1: K=64(padded), N=288 -> f16
  k_gemm_h<KPAD, false, true><<<dim3((unsigned)(MP / 64), Ec / 48), 128, 0, stream>>>(
      tok_h, w_lb1, lb1_b, nullptr, e_h, (int)MP, Ec);
  // qkv1: K=288, N=864 -> f16
  k_gemm_h<Ec, false, true><<<dim3((unsigned)(MP / 64), 3 * Ec / 48), 128, 0, stream>>>(
      e_h, w_qkv1, n1_qkvb, nullptr, qkv_h, (int)MP, 3 * Ec);
  k_nat<<<(unsigned)((Bc * Hc * Wc * NHh + 255) / 256), 256, 0, stream>>>(qkv_h, attn_h, Hc, Wc);
  // proj1: K=288, N=288 -> f16
  k_gemm_h<Ec, false, true><<<dim3((unsigned)(MP / 64), Ec / 48), 128, 0, stream>>>(
      attn_h, w_pj1, n1_pb, nullptr, proj_h, (int)MP, Ec);
  // la1: K=288, N=48 -> f32
  k_gemm_h<Ec, true, false><<<dim3((unsigned)(MP / 64), 1), 128, 0, stream>>>(
      proj_h, w_la1, la1_b, xp_fin, nullptr, (int)MP, 48);

  // ================= region branch =================
  k_down<<<(unsigned)MR, 48, 0, stream>>>(x, ln2_w, ln2_b, cdw, cdb, xr0);
  k_ln_inproj<<<(unsigned)MR, 96, 0, stream>>>(xr0, ln1_w, ln1_b, 0, mr_inw, xs_raw, z_raw, L2c);
  {
    size_t tot = (size_t)Bc * 48 * L2c;
    unsigned g = (unsigned)((tot + 255) / 256);
    k_dwconv_silu<<<g, 256, 0, stream>>>(xs_raw, mr_cxw, xs_c, L2c);
    k_dwconv_silu<<<g, 256, 0, stream>>>(z_raw, mr_czw, z_c, L2c);
  }
  k_xproj<<<(unsigned)MR, 48, 0, stream>>>(xs_c, mr_xpw, mr_dtw, dtbuf, Bm, Cm, L2c);
  k_scan<<<Bc * 48, 32, 0, stream>>>(xs_c, dtbuf, mr_alog, mr_D, mr_dtb, Bm, Cm, ybuf, L2c);
  k_outproj<<<(unsigned)MR, 48, 0, stream>>>(ybuf, z_c, mr_opw, tok_h, L2c);
  k_gemm_h<KPAD, false, true><<<dim3((unsigned)(MR / 64), Ec / 48), 128, 0, stream>>>(
      tok_h, w_lb2, lb2_b, nullptr, e_h, (int)MR, Ec);
  k_gemm_h<Ec, false, true><<<dim3((unsigned)(MR / 64), 3 * Ec / 48), 128, 0, stream>>>(
      e_h, w_qkv2, n2_qkvb, nullptr, qkv_h, (int)MR, 3 * Ec);
  k_nat<<<(unsigned)((Bc * H2c * W2c * NHh + 255) / 256), 256, 0, stream>>>(qkv_h, attn_h, H2c, W2c);
  k_gemm_h<Ec, false, true><<<dim3((unsigned)(MR / 64), Ec / 48), 128, 0, stream>>>(
      attn_h, w_pj2, n2_pb, nullptr, proj_h, (int)MR, Ec);
  k_gemm_h<Ec, true, false><<<dim3((unsigned)(MR / 64), 1), 128, 0, stream>>>(
      proj_h, w_la2, la2_b, xr_tok, nullptr, (int)MR, 48);
  k_up<<<(unsigned)MP, 48, 0, stream>>>(xr_tok, cbw, cbb, caw, cab, xr_img);

  // ================= fusion =================
  k_conv1<<<(unsigned)((MP * 48 + 255) / 256), 256, 0, stream>>>(xp_fin, xr_img, c1w, c1b, gbuf);
  k_fuse<<<(unsigned)MP, 48, 0, stream>>>(gbuf, c2w, c2b, xp_fin, xr_img, lw, lbv, x, out);
}
